// KANConv2d_46084999086776
// MI455X (gfx1250) — matrix operand order repaired and checked by reading, not compiled, not run on hardware
//
#include <hip/hip_runtime.h>
#include <hip/hip_bf16.h>
#include <cstdint>

typedef _Float16 v16h __attribute__((ext_vector_type(16)));
typedef _Float16 h8   __attribute__((ext_vector_type(8)));
typedef float    v8f  __attribute__((ext_vector_type(8)));
typedef int      v4i  __attribute__((ext_vector_type(4)));

#define IN_CH   32
#define OUT_CH  64
#define HW      64
#define COEF    8
#define VALS    9          // silu + 8 spline coefficients
#define KSTEPS  81         // 9 vals * 9 taps ; K-total = 81*32 = 2592
#define TILE    16         // 16x16 spatial tile
#define HALO    (TILE + 2) // 18

// ---------------------------------------------------------------------------
// Async global->LDS staging (CDNA5 GLOBAL_LOAD_ASYNC_TO_LDS_B128, ASYNCcnt),
// with a synchronous fallback if the builtin is not declared.
// Builtin signature (from hipcc diagnostic): param0 = int4 addrspace(1)*,
// param1 = int4 addrspace(3)*, then two immediate ints (offset, cpol).
// ---------------------------------------------------------------------------
__device__ __forceinline__ void stage_b128(const _Float16* g, _Float16* l) {
#if defined(__AMDGCN__) && __has_builtin(__builtin_amdgcn_global_load_async_to_lds_b128)
    __builtin_amdgcn_global_load_async_to_lds_b128(
        (__attribute__((address_space(1))) v4i*)g,
        (__attribute__((address_space(3))) v4i*)l, 0, 0);
#else
    *(uint4*)l = *(const uint4*)g;
#endif
}

__device__ __forceinline__ void wait_stage() {
#if defined(__AMDGCN__) && __has_builtin(__builtin_amdgcn_global_load_async_to_lds_b128)
#if __has_builtin(__builtin_amdgcn_s_wait_asynccnt)
    __builtin_amdgcn_s_wait_asynccnt(0);
#else
    asm volatile("s_wait_asynccnt 0x0" ::: "memory");
#endif
#endif
}

// ---------------------------------------------------------------------------
// Pack base_weight [64,288] + spline_weight [64,288,8] (f32) into a per-lane
// v_wmma_f32_16x16x32_f16 B-operand layout (f16, contiguous k order, same as A):
//   Bpack[s][ntile][lane][idx]; s = v*9 + t; lane: n = ntile*16 + lane%16,
//   khalf = lane/16; within-step K (=channel) c = 16*khalf + idx.
//   unfold feature f = c*9 + t (channel-major, tap-minor).
// ---------------------------------------------------------------------------
__global__ __launch_bounds__(256) void kan_pack_b(const float* __restrict__ bw,
                                                  const float* __restrict__ sw,
                                                  _Float16* __restrict__ bp) {
    int gid = blockIdx.x * 256 + threadIdx.x;          // 81*4*32*16 = 165888
    if (gid >= KSTEPS * 4 * 32 * 16) return;
    int idx  = gid & 15;
    int lane = (gid >> 4) & 31;
    int nt   = (gid >> 9) & 3;
    int s    = gid >> 11;
    int v = s / 9, t = s % 9;
    int c = 16 * (lane >> 4) + idx;
    int n = nt * 16 + (lane & 15);
    int f = c * 9 + t;
    float val = (v == 0) ? bw[n * 288 + f]
                         : sw[(n * 288 + f) * COEF + (v - 1)];
    bp[gid] = (_Float16)val;
}

// ---------------------------------------------------------------------------
// Fused KAN-conv: expand halo once into LDS, then K=2592 WMMA GEMM with
// LDS-staged, double-buffered, async-loaded B. 256 threads = 8 waves;
// spatial tile 16x16; wave w owns output rows 2w, 2w+1 (2 M-tiles).
// ---------------------------------------------------------------------------
__global__ __launch_bounds__(256) void kan_conv_wmma(const float* __restrict__ x,
                                                     const _Float16* __restrict__ Bpack,
                                                     float* __restrict__ out) {
    __shared__ union {
        _Float16 expv[VALS][HALO][HALO][IN_CH];  // 186624 B (phases 1-2)
        float    outs[TILE][OUT_CH][17];         //  69632 B (phase 3, padded)
    } sm;
    __shared__ _Float16 Bs[2][4 * 32 * 16];      //   8192 B double buffer

    const int b   = blockIdx.x;                  // 16 imgs * 4 * 4 tiles = 256
    const int n   = b >> 4;
    const int iy0 = ((b >> 2) & 3) * TILE;
    const int ix0 = (b & 3) * TILE;
    const int tid = threadIdx.x;

    // Kick off async staging of B k-step 0 (overlaps with expansion phase).
    stage_b128(Bpack + tid * 8, &Bs[0][tid * 8]);

    // ---- Phase 1: expansion of 18x18x32 halo (silu + cubic B-spline) ----
    const float h = 0.4f;
    for (int e = tid; e < HALO * HALO * IN_CH; e += 256) {
        int c  = e / (HALO * HALO);
        int r  = e - c * (HALO * HALO);
        int ly = r / HALO, lx = r - ly * HALO;
        int gy = iy0 - 1 + ly, gx = ix0 - 1 + lx;
        float xv = 0.0f;                         // zero pad (reference pads pre-basis)
        if (gy >= 0 && gy < HW && gx >= 0 && gx < HW)
            xv = x[(((size_t)n * IN_CH + c) * HW + gy) * HW + gx];

        float sil = xv / (1.0f + __expf(-xv));

        float bb[11];                            // knots t_i = (i-3)*0.4 - 1
#pragma unroll
        for (int i = 0; i < 11; ++i) {
            float t0 = h * (float)(i - 3) - 1.0f;
            bb[i] = (xv >= t0 && xv < t0 + h) ? 1.0f : 0.0f;
        }
#pragma unroll
        for (int k = 1; k <= 3; ++k) {
            float inv = 1.0f / (h * (float)k);
#pragma unroll
            for (int i = 0; i < 10; ++i) {
                if (i < 11 - k) {
                    float ti   = h * (float)(i - 3) - 1.0f;
                    float tik1 = h * (float)(i + k - 2) - 1.0f;
                    bb[i] = (xv - ti) * inv * bb[i] + (tik1 - xv) * inv * bb[i + 1];
                }
            }
        }
        sm.expv[0][ly][lx][c] = (_Float16)sil;
#pragma unroll
        for (int i = 0; i < COEF; ++i)
            sm.expv[1 + i][ly][lx][c] = (_Float16)bb[i];
    }
    wait_stage();
    __syncthreads();

    // ---- Phase 2: 81 K-steps of WMMA; B double-buffered through LDS ----
    const int w     = tid >> 5;                  // wave id -> rows 2w, 2w+1
    const int lane  = tid & 31;
    const int m     = lane & 15;                 // M (spatial column)
    const int khalf = lane >> 4;

    v8f acc[2][4] = {};

    for (int v = 0; v < VALS; ++v) {
#pragma unroll
        for (int t = 0; t < 9; ++t) {
            const int s   = v * 9 + t;
            const int dy  = t / 3, dx = t % 3;
            const int cur = s & 1;
            if (s < KSTEPS - 1)                  // prefetch next B slice (4 KB)
                stage_b128(Bpack + (size_t)(s + 1) * 2048 + tid * 8,
                           &Bs[cur ^ 1][tid * 8]);

            v16h a[2];
#pragma unroll
            for (int rr = 0; rr < 2; ++rr) {
                const _Float16* ap = &sm.expv[v][2 * w + rr + dy][m + dx][0];
                h8 lo = *(const h8*)(ap + 16 * khalf);
                h8 hi = *(const h8*)(ap + 16 * khalf + 8);
#pragma unroll
                for (int i = 0; i < 8; ++i) { a[rr][i] = lo[i]; a[rr][8 + i] = hi[i]; }
            }
#pragma unroll
            for (int nt = 0; nt < 4; ++nt) {
                v16h bf = *(const v16h*)(&Bs[cur][(nt * 32 + lane) * 16]);
                acc[0][nt] = __builtin_amdgcn_wmma_f32_16x16x32_f16(
                    false, a[0], false, bf, (short)0, acc[0][nt], false, false);
                acc[1][nt] = __builtin_amdgcn_wmma_f32_16x16x32_f16(
                    false, a[1], false, bf, (short)0, acc[1][nt], false, false);
            }
            wait_stage();
            __syncthreads();
        }
    }

    // ---- Phase 3: transpose D through LDS, then coalesced b128 stores ----
#pragma unroll
    for (int rr = 0; rr < 2; ++rr)
#pragma unroll
        for (int nt = 0; nt < 4; ++nt)
#pragma unroll
            for (int r = 0; r < 8; ++r)
                sm.outs[2 * w + rr][nt * 16 + m][r + 8 * khalf] = acc[rr][nt][r];
    __syncthreads();

    const int o = tid >> 2, q = tid & 3;         // 64 channels x 4 quads
#pragma unroll 4
    for (int yl = 0; yl < TILE; ++yl) {
        float4 vv;
        vv.x = sm.outs[yl][o][4 * q + 0];
        vv.y = sm.outs[yl][o][4 * q + 1];
        vv.z = sm.outs[yl][o][4 * q + 2];
        vv.w = sm.outs[yl][o][4 * q + 3];
        *(float4*)(out + (((size_t)n * OUT_CH + o) * HW + iy0 + yl) * HW + ix0 + 4 * q) = vv;
    }
}

extern "C" void kernel_launch(void* const* d_in, const int* in_sizes, int n_in,
                              void* d_out, int out_size, void* d_ws, size_t ws_size,
                              hipStream_t stream) {
    const float* x  = (const float*)d_in[0];   // [16,32,64,64]
    const float* bw = (const float*)d_in[1];   // [64,288]
    const float* sw = (const float*)d_in[2];   // [64,288,8]
    float* out      = (float*)d_out;           // [16,64,64,64]
    _Float16* Bpack = (_Float16*)d_ws;         // 331776 bytes used

    kan_pack_b<<<(KSTEPS * 4 * 32 * 16 + 255) / 256, 256, 0, stream>>>(bw, sw, Bpack);
    kan_conv_wmma<<<256, 256, 0, stream>>>(x, Bpack, out);
}
